// PointNetEncoder_90494960927419
// MI455X (gfx1250) — compile-verified, matrix operand order checked by reading, not run on hardware
//
#include <hip/hip_runtime.h>
#include <hip/hip_bf16.h>

// ---------------------------------------------------------------------------
// PointNetEncoder (3->64->128->1024 conv+BN, histogram-argmax feature)
// CDNA5 / gfx1250 wave32 implementation.
//
// Pipeline (all intermediates fit in the 192MB L2):
//  1) statsx  : Sx[3], Sxx[3x3] of input            -> analytic BN1 stats
//  2) fin1    : fold BN1 into W1e[64x3], b1e
//  3) gram1   : h1 on-the-fly, s1[64] + G1[64x64] via v_wmma bf16
//  4) fin2    : analytic BN2 stats from (s1,G1), fold -> W2e bf16, b2e
//  5) h2      : h2 = relu(W2e·h1 + b2e) via WMMA, store bf16 [16,4096,128] (n-major)
//  6) gram2   : s2[128] + G2[128x128] of h2 via WMMA (TDM-staged tiles)
//  7) fin3    : analytic BN3 stats from (s2,G2), fold -> W3e bf16, b3e
//  8) hist    : per (b, 16-ch tile): y3 tile via WMMA on TDM double-buffered
//               LDS tiles; bin into LDS hist, leak + merge into flat histogram
//  9) argmax  : first-max bin index per group -> d_out
// ---------------------------------------------------------------------------

typedef __attribute__((ext_vector_type(16))) __bf16 v16bf;
typedef __attribute__((ext_vector_type(8)))  float  v8f;
typedef __attribute__((ext_vector_type(4)))  unsigned tdm_v4u;
typedef __attribute__((ext_vector_type(8)))  int      tdm_v8i;
typedef __attribute__((ext_vector_type(4)))  int      tdm_v4i;

#define Bsz   16
#define Npts  4096
#define Qtot  65536            // B*N
#define BINS  70
#define NGRP  16384            // B * 1024
#define TOTAL (BINS * NGRP)    // 1146880

// ---- workspace layout (float units) ----
#define OF_SX    0        // 3  (use 4)
#define OF_SXX   4        // 6  (use 12)
#define OF_S1    16       // 64
#define OF_S2    80       // 128
#define OF_G1    256      // 4096
#define OF_G2    4352     // 16384
#define ACC_END  20736    // zeroed region [0, ACC_END)
#define OF_W1E   20736    // 192
#define OF_B1E   20928    // 64
#define OF_B2E   20992    // 128
#define OF_B3E   21120    // 1024
#define OF_W2E   22272    // 8192 bf16 = 4096 floats
#define OF_W3E   26368    // 131072 bf16 = 65536 floats
#define OF_FLAT  91904    // 1146880 u32
#define OF_H2    1239040  // 8388608 bf16 = 4194304 floats

// ---------------------------------------------------------------------------
// WMMA helpers: ISA lane layout for 16-bit 16x32 A / 32x16 B matrices.
// lane l: half h = l>>4.  element j of v16bf maps to
//   K(j) = (j<8) ? h*8+j : 16 + h*8 + (j-8)
// A: row M = l&15, elem A[M][K(j)].   B: col N = l&15, elem B[K(j)][N].
// C/D: VGPR r of lane l holds (M = r + 8*h, N = l&15).
// ---------------------------------------------------------------------------
__device__ __forceinline__ int wmma_k(int half, int j) {
  return (j < 8) ? (half * 8 + j) : (16 + half * 8 + (j - 8));
}

// A[m][k] = lds[(mbase+m)*pitch + kbase + k] -- contiguous per lane (b128-able)
__device__ __forceinline__ v16bf load_A(const __bf16* lds, int pitch,
                                        int mbase, int kbase) {
  int l = threadIdx.x & 31, h = l >> 4, m = l & 15;
  v16bf a;
#pragma unroll
  for (int j = 0; j < 16; ++j)
    a[j] = lds[(mbase + m) * pitch + kbase + wmma_k(h, j)];
  return a;
}

// B[k][n] = lds[(kbase+k)*pitch + nbase + n] -- K strided across rows
__device__ __forceinline__ v16bf load_B(const __bf16* lds, int pitch,
                                        int kbase, int nbase) {
  int l = threadIdx.x & 31, h = l >> 4, n = l & 15;
  v16bf b;
#pragma unroll
  for (int j = 0; j < 16; ++j)
    b[j] = lds[(kbase + wmma_k(h, j)) * pitch + nbase + n];
  return b;
}

__device__ __forceinline__ v8f wmma_bf16(v16bf a, v16bf b, v8f c) {
  return __builtin_amdgcn_wmma_f32_16x16x32_bf16(false, a, false, b,
                                                 (short)0, c, false, false);
}

// ---------------------------------------------------------------------------
// Tensor Data Mover: 2D tile (16-bit elems) global -> LDS.
//   dim0/tile0 : contiguous dimension (elements), row stride = stride0 elems.
//   LDS rows padded: pad_int code (5 => every 64 DWORDs), pad_amt DWORDs.
// Issue from ONE wave per block; completion via TENSORcnt (in-order per wave).
// This toolchain exposes the 6-arg builtin: (g0, g1, g2, g3, g4, cpol);
// 2D tensors leave descriptor groups 2/3 (and the trailing group) zeroed.
// ---------------------------------------------------------------------------
__device__ __forceinline__ void tdm_load_2d(unsigned lds_addr,
                                            unsigned long long gaddr,
                                            unsigned dim0, unsigned dim1,
                                            unsigned tile0, unsigned tile1,
                                            unsigned stride0,
                                            unsigned pad_int, unsigned pad_amt) {
  tdm_v4u g0;
  g0[0] = 1u;                                    // count=1, user descriptor
  g0[1] = lds_addr;                              // LDS byte address
  g0[2] = (unsigned)(gaddr & 0xffffffffull);     // global_addr[31:0]
  g0[3] = (unsigned)((gaddr >> 32) & 0x01ffffffull) | (2u << 30);  // type=2
  tdm_v8i g1;
  unsigned w0 = (1u << 16);                      // data_size=1 (2 bytes)
  if (pad_amt) w0 |= (1u << 20) | (pad_int << 22) | ((pad_amt - 1u) << 25);
  g1[0] = (int)w0;
  g1[1] = (int)((dim0 & 0xffffu) << 16);                 // tensor_dim0 lo16
  g1[2] = (int)((dim0 >> 16) | ((dim1 & 0xffffu) << 16));
  g1[3] = (int)((dim1 >> 16) | (tile0 << 16));
  g1[4] = (int)(tile1 & 0xffffu);                        // tile_dim1, dim2=0
  g1[5] = (int)stride0;                                  // dim0 stride lo32
  g1[6] = 0;
  g1[7] = 0;
  tdm_v4i z4 = {0, 0, 0, 0};
  tdm_v8i z8 = {0, 0, 0, 0, 0, 0, 0, 0};
  __builtin_amdgcn_tensor_load_to_lds(g0, g1, z4, z4, z8, 0);
}

__device__ __forceinline__ unsigned lds_addr_of(const void* p) {
  return (unsigned)(unsigned long long)(uintptr_t)p;
}

// ---------------------------------------------------------------------------
__global__ __launch_bounds__(256) void k_zero(unsigned* p, int n) {
  int i = blockIdx.x * 256 + threadIdx.x;
  if (i < n) p[i] = 0u;
}

// ---- 1) input first/second moments --------------------------------------
__global__ __launch_bounds__(256) void k_statsx(const float* __restrict__ x,
                                                float* sx, float* sxx) {
  int b = blockIdx.x >> 2, n0 = (blockIdx.x & 3) * 1024;
  float a[9] = {0.f, 0.f, 0.f, 0.f, 0.f, 0.f, 0.f, 0.f, 0.f};
  for (int t = threadIdx.x; t < 1024; t += 256) {
    int n = n0 + t;
    float x0 = x[(b * 3 + 0) * Npts + n];
    float x1 = x[(b * 3 + 1) * Npts + n];
    float x2 = x[(b * 3 + 2) * Npts + n];
    a[0] += x0; a[1] += x1; a[2] += x2;
    a[3] += x0 * x0; a[4] += x0 * x1; a[5] += x0 * x2;
    a[6] += x1 * x1; a[7] += x1 * x2; a[8] += x2 * x2;
  }
  __shared__ float red[256];
  for (int q = 0; q < 9; ++q) {
    red[threadIdx.x] = a[q];
    __syncthreads();
    for (int s = 128; s > 0; s >>= 1) {
      if (threadIdx.x < s) red[threadIdx.x] += red[threadIdx.x + s];
      __syncthreads();
    }
    if (threadIdx.x == 0) {
      if (q < 3) atomicAdd(&sx[q], red[0]);
      else       atomicAdd(&sxx[q - 3], red[0]);
    }
    __syncthreads();
  }
}

// ---- 2) fold BN1 analytically -------------------------------------------
__global__ void k_fin1(const float* w1, const float* b1, const float* g1,
                       const float* be1, const float* sx, const float* sxx,
                       float* W1e, float* b1e) {
  int o = threadIdx.x;
  if (o >= 64) return;
  const float invQ = 1.0f / (float)Qtot;
  float w0 = w1[o * 3], wa = w1[o * 3 + 1], wb = w1[o * 3 + 2], bb = b1[o];
  float mx0 = sx[0] * invQ, mx1 = sx[1] * invQ, mx2 = sx[2] * invQ;
  float wm = w0 * mx0 + wa * mx1 + wb * mx2;
  float mean = wm + bb;
  float q = w0 * w0 * sxx[0] + 2.f * w0 * wa * sxx[1] + 2.f * w0 * wb * sxx[2]
          + wa * wa * sxx[3] + 2.f * wa * wb * sxx[4] + wb * wb * sxx[5];
  float e2 = q * invQ + 2.f * bb * wm + bb * bb;
  float var = e2 - mean * mean;
  float aa = g1[o] * rsqrtf(var + 1e-5f);
  float cc = be1[o] - mean * aa;
  W1e[o * 3 + 0] = aa * w0;
  W1e[o * 3 + 1] = aa * wa;
  W1e[o * 3 + 2] = aa * wb;
  b1e[o] = aa * bb + cc;
}

// ---- 3) Gram of h1 (h1 computed on the fly) ------------------------------
__global__ __launch_bounds__(256) void k_gram1(const float* __restrict__ x,
                                               const float* W1e,
                                               const float* b1e,
                                               float* s1, float* G1) {
  int b = blockIdx.x >> 2, n0 = (blockIdx.x & 3) * 1024;
  __shared__ __bf16 h1l[64 * 40];
  int w = threadIdx.x >> 5, lane = threadIdx.x & 31;
  int Mc0 = w >> 2, Nc0 = w & 3;
  int Mc1 = (w + 8) >> 2, Nc1 = (w + 8) & 3;
  v8f acc0 = {0.f, 0.f, 0.f, 0.f, 0.f, 0.f, 0.f, 0.f};
  v8f acc1 = acc0;
  float s1acc = 0.f;
  for (int it = 0; it < 32; ++it) {
    int nc = n0 + it * 32;
    for (int i = threadIdx.x; i < 64 * 32; i += 256) {
      int c = i >> 5, nn = i & 31, n = nc + nn;
      float x0 = x[(b * 3 + 0) * Npts + n];
      float x1 = x[(b * 3 + 1) * Npts + n];
      float x2 = x[(b * 3 + 2) * Npts + n];
      float v = W1e[c * 3] * x0 + W1e[c * 3 + 1] * x1 + W1e[c * 3 + 2] * x2 +
                b1e[c];
      h1l[c * 40 + nn] = (__bf16)fmaxf(v, 0.f);
    }
    __syncthreads();
    if (threadIdx.x < 64) {
#pragma unroll 8
      for (int j = 0; j < 32; ++j) s1acc += (float)h1l[threadIdx.x * 40 + j];
    }
    v16bf aA = load_A(h1l, 40, Mc0 * 16, 0);
    v16bf bB = load_A(h1l, 40, Nc0 * 16, 0);
    acc0 = wmma_bf16(aA, bB, acc0);
    aA = load_A(h1l, 40, Mc1 * 16, 0);
    bB = load_A(h1l, 40, Nc1 * 16, 0);
    acc1 = wmma_bf16(aA, bB, acc1);
    __syncthreads();
  }
  if (threadIdx.x < 64) atomicAdd(&s1[threadIdx.x], s1acc);
#pragma unroll
  for (int r = 0; r < 8; ++r) {
    int M = r + (lane >> 4) * 8, Nn = lane & 15;
    atomicAdd(&G1[(Mc0 * 16 + M) * 64 + Nc0 * 16 + Nn], acc0[r]);
    atomicAdd(&G1[(Mc1 * 16 + M) * 64 + Nc1 * 16 + Nn], acc1[r]);
  }
}

// ---- 4) fold BN2 ---------------------------------------------------------
__global__ void k_fin2(const float* w2, const float* b2, const float* g2,
                       const float* be2, const float* s1, const float* G1,
                       __bf16* W2e, float* b2e) {
  int o = threadIdx.x;
  if (o >= 128) return;
  const float invQ = 1.0f / (float)Qtot;
  const float* w = w2 + o * 64;
  float dot = 0.f;
  for (int c = 0; c < 64; ++c) dot += w[c] * s1[c];
  float q = 0.f;
  for (int c = 0; c < 64; ++c) {
    const float* g = G1 + c * 64;
    float t = 0.f;
    for (int d = 0; d < 64; ++d) t += g[d] * w[d];
    q += w[c] * t;
  }
  float bb = b2[o];
  float mean = dot * invQ + bb;
  float e2 = q * invQ + 2.f * bb * dot * invQ + bb * bb;
  float var = e2 - mean * mean;
  float aa = g2[o] * rsqrtf(var + 1e-5f);
  float cc = be2[o] - mean * aa;
  for (int c = 0; c < 64; ++c) W2e[o * 64 + c] = (__bf16)(aa * w[c]);
  b2e[o] = aa * bb + cc;
}

// ---- 5) h2 = relu(W2e·h1 + b2e) via WMMA, store bf16 n-major -------------
// h2T layout: [b][n][c], c contiguous (128). Each lane's 8 C-matrix values are
// 8 consecutive channels at fixed n -> one aligned 16B store.
__global__ __launch_bounds__(256) void k_h2(const float* __restrict__ x,
                                            const float* W1e, const float* b1e,
                                            const __bf16* __restrict__ W2e,
                                            const float* b2e,
                                            __bf16* __restrict__ h2) {
  int b = blockIdx.x >> 4, n0 = (blockIdx.x & 15) * 256;
  __shared__ __bf16 w2l[128 * 68];
  __shared__ __bf16 h1l[64 * 40];
  int w = threadIdx.x >> 5, lane = threadIdx.x & 31;
  for (int i = threadIdx.x; i < 128 * 64; i += 256)
    w2l[(i >> 6) * 68 + (i & 63)] = W2e[i];
  __syncthreads();
  for (int it = 0; it < 8; ++it) {
    int nc = n0 + it * 32;
    for (int i = threadIdx.x; i < 64 * 32; i += 256) {
      int c = i >> 5, nn = i & 31, n = nc + nn;
      float x0 = x[(b * 3 + 0) * Npts + n];
      float x1 = x[(b * 3 + 1) * Npts + n];
      float x2 = x[(b * 3 + 2) * Npts + n];
      float v = W1e[c * 3] * x0 + W1e[c * 3 + 1] * x1 + W1e[c * 3 + 2] * x2 +
                b1e[c];
      h1l[c * 40 + nn] = (__bf16)fmaxf(v, 0.f);
    }
    __syncthreads();
    v16bf a0 = load_A(w2l, 68, w * 16, 0);
    v16bf a1 = load_A(w2l, 68, w * 16, 32);
#pragma unroll
    for (int ns = 0; ns < 2; ++ns) {
      v16bf bm0 = load_B(h1l, 40, 0, ns * 16);
      v16bf bm1 = load_B(h1l, 40, 32, ns * 16);
      v8f acc = {0.f, 0.f, 0.f, 0.f, 0.f, 0.f, 0.f, 0.f};
      acc = wmma_bf16(a0, bm0, acc);
      acc = wmma_bf16(a1, bm1, acc);
      int n = nc + ns * 16 + (lane & 15);
      int cbase = w * 16 + (lane >> 4) * 8;
      union { __bf16 h[8]; uint4 u; } pk;
#pragma unroll
      for (int r = 0; r < 8; ++r) {
        float v = acc[r] + b2e[cbase + r];
        pk.h[r] = (__bf16)fmaxf(v, 0.f);
      }
      *(uint4*)(h2 + ((size_t)(b * Npts + n)) * 128 + cbase) = pk.u;
    }
    __syncthreads();
  }
}

// ---- 6) Gram of h2 (TDM double-buffered tiles) ---------------------------
__global__ __launch_bounds__(256) void k_gram2(const __bf16* __restrict__ h2,
                                               float* s2, float* G2) {
  int b = blockIdx.x >> 2, n0 = (blockIdx.x & 3) * 1024;
  __shared__ __bf16 h2l[2][32 * 136];
  int tid = threadIdx.x, w = tid >> 5, lane = tid & 31;
  bool w0 = (tid < 32);
  v8f acc[8] = {};
  float sacc = 0.f;
  unsigned long long hbase = (unsigned long long)(uintptr_t)h2;
  unsigned lds[2] = {lds_addr_of(&h2l[0][0]), lds_addr_of(&h2l[1][0])};
  if (w0)
    tdm_load_2d(lds[0], hbase + 2ull * ((unsigned long long)(b * Npts + n0)) * 128ull,
                128, 32, 128, 32, 128, 5, 4);
  for (int it = 0; it < 32; ++it) {
    if (w0) {
      if (it + 1 < 32) {
        tdm_load_2d(lds[(it + 1) & 1],
                    hbase + 2ull * ((unsigned long long)(b * Npts + n0 + (it + 1) * 32)) * 128ull,
                    128, 32, 128, 32, 128, 5, 4);
        __builtin_amdgcn_s_wait_tensorcnt(1);
      } else {
        __builtin_amdgcn_s_wait_tensorcnt(0);
      }
    }
    __syncthreads();
    const __bf16* cur = &h2l[it & 1][0];
    if (tid < 128) {
#pragma unroll 8
      for (int nr = 0; nr < 32; ++nr) sacc += (float)cur[nr * 136 + tid];
    }
    // Gram tiles on n-major data: element[k=n][c] -> load_B pattern
    v16bf aA = load_B(cur, 136, 0, w * 16);
#pragma unroll
    for (int i = 0; i < 8; ++i) {
      v16bf bB = load_B(cur, 136, 0, i * 16);
      acc[i] = wmma_bf16(aA, bB, acc[i]);
    }
    __syncthreads();
  }
  if (tid < 128) atomicAdd(&s2[tid], sacc);
#pragma unroll
  for (int i = 0; i < 8; ++i)
#pragma unroll
    for (int r = 0; r < 8; ++r) {
      int M = r + (lane >> 4) * 8, Nn = lane & 15;
      atomicAdd(&G2[(w * 16 + M) * 128 + i * 16 + Nn], acc[i][r]);
    }
}

// ---- 7) fold BN3 (into bf16 W3e rows) ------------------------------------
__global__ __launch_bounds__(256) void k_fin3(const float* w3, const float* b3,
                                              const float* g3, const float* be3,
                                              const float* s2, const float* G2,
                                              __bf16* W3e, float* b3e) {
  int o = blockIdx.x * 256 + threadIdx.x;
  if (o >= 1024) return;
  const float invQ = 1.0f / (float)Qtot;
  const float* w = w3 + o * 128;
  float dot = 0.f;
  for (int c = 0; c < 128; ++c) dot += w[c] * s2[c];
  float q = 0.f;
  for (int c = 0; c < 128; ++c) {
    const float* g = G2 + c * 128;
    float t = 0.f;
    for (int d = 0; d < 128; ++d) t += g[d] * w[d];
    q += w[c] * t;
  }
  float bb = b3[o];
  float mean = dot * invQ + bb;
  float e2 = q * invQ + 2.f * bb * dot * invQ + bb * bb;
  float var = e2 - mean * mean;
  float aa = g3[o] * rsqrtf(var + 1e-5f);
  float cc = be3[o] - mean * aa;
  for (int c = 0; c < 128; ++c) W3e[o * 128 + c] = (__bf16)(aa * w[c]);
  b3e[o] = aa * bb + cc;  // normalized y3 = W3e·h2 + b3e (no relu)
}

// ---- 8) histogram: y3 tile via WMMA on TDM double-buffered tiles ---------
__global__ __launch_bounds__(256) void k_hist(const __bf16* __restrict__ h2,
                                              const __bf16* __restrict__ W3e,
                                              const float* b3e,
                                              unsigned* __restrict__ flat) {
  int b = blockIdx.x >> 6, ot = blockIdx.x & 63;
  __shared__ unsigned hist[16 * 72];
  __shared__ __bf16 w3l[16 * 136];
  __shared__ __bf16 h2l[2][128 * 136];
  int tid = threadIdx.x, w = tid >> 5, lane = tid & 31;
  bool w0 = (tid < 32);
  for (int i = tid; i < 16 * 72; i += 256) hist[i] = 0u;
  for (int i = tid; i < 16 * 128; i += 256) {
    int row = i >> 7, col = i & 127;
    w3l[row * 136 + col] = W3e[(ot * 16 + row) * 128 + col];
  }
  unsigned long long hbase = (unsigned long long)(uintptr_t)h2;
  unsigned lds[2] = {lds_addr_of(&h2l[0][0]), lds_addr_of(&h2l[1][0])};
  if (w0)
    tdm_load_2d(lds[0], hbase + 2ull * ((unsigned long long)b * Npts) * 128ull,
                128, 128, 128, 128, 128, 5, 4);
  int grp0 = b * 1024 + ot * 16;
  for (int ch = 0; ch < 32; ++ch) {
    if (w0) {
      if (ch + 1 < 32) {
        tdm_load_2d(lds[(ch + 1) & 1],
                    hbase + 2ull * ((unsigned long long)(b * Npts + (ch + 1) * 128)) * 128ull,
                    128, 128, 128, 128, 128, 5, 4);
        __builtin_amdgcn_s_wait_tensorcnt(1);
      } else {
        __builtin_amdgcn_s_wait_tensorcnt(0);
      }
    }
    __syncthreads();
    const __bf16* cur = &h2l[ch & 1][0];
    v8f acc = {0.f, 0.f, 0.f, 0.f, 0.f, 0.f, 0.f, 0.f};
#pragma unroll
    for (int k = 0; k < 4; ++k) {
      v16bf aA = load_A(w3l, 136, 0, k * 32);
      // B[k=c][n] on n-major tile = cur[n*136 + c] -> load_A pattern (b128)
      v16bf bB = load_A(cur, 136, w * 16, k * 32);
      acc = wmma_bf16(aA, bB, acc);
    }
#pragma unroll
    for (int r = 0; r < 8; ++r) {
      int M = r + (lane >> 4) * 8;
      float v = acc[r] + b3e[ot * 16 + M];
      float fb = floorf((v + 10.0f) * 3.5f);  // (v - HLO) / (20/70)
      if (fb >= 0.0f && fb < 70.0f) {
        atomicAdd(&hist[M * 72 + (int)fb], 1u);
      } else {  // leak into neighboring groups (global-offset hack)
        float ff = fb + 70.0f * (float)(grp0 + M);
        if (ff >= 0.0f && ff < (float)TOTAL) atomicAdd(&flat[(int)ff], 1u);
      }
    }
    __syncthreads();
  }
  for (int i = tid; i < 16 * 70; i += 256) {
    int row = i / 70, j = i - row * 70;
    unsigned cnt = hist[row * 72 + j];
    if (cnt) atomicAdd(&flat[(grp0 + row) * 70 + j], cnt);
  }
}

// ---- 9) argmax (first max) ----------------------------------------------
__global__ __launch_bounds__(256) void k_argmax(const unsigned* __restrict__ flat,
                                                float* __restrict__ out) {
  int g = blockIdx.x * 256 + threadIdx.x;
  if (g >= NGRP) return;
  const unsigned* h = flat + g * BINS;
  unsigned best = h[0];
  int bi = 0;
  for (int j = 1; j < BINS; ++j) {
    unsigned v = h[j];
    if (v > best) { best = v; bi = j; }
  }
  out[g] = (float)bi;
}

// ---------------------------------------------------------------------------
extern "C" void kernel_launch(void* const* d_in, const int* in_sizes, int n_in,
                              void* d_out, int out_size, void* d_ws,
                              size_t ws_size, hipStream_t stream) {
  (void)in_sizes; (void)n_in; (void)out_size; (void)ws_size;
  const float* x   = (const float*)d_in[0];
  const float* w1  = (const float*)d_in[1];
  const float* b1  = (const float*)d_in[2];
  const float* g1  = (const float*)d_in[3];
  const float* be1 = (const float*)d_in[4];
  const float* w2  = (const float*)d_in[5];
  const float* b2  = (const float*)d_in[6];
  const float* g2  = (const float*)d_in[7];
  const float* be2 = (const float*)d_in[8];
  const float* w3  = (const float*)d_in[9];
  const float* b3  = (const float*)d_in[10];
  const float* g3  = (const float*)d_in[11];
  const float* be3 = (const float*)d_in[12];

  float* W = (float*)d_ws;
  float*    sx   = W + OF_SX;
  float*    sxx  = W + OF_SXX;
  float*    s1   = W + OF_S1;
  float*    s2   = W + OF_S2;
  float*    G1   = W + OF_G1;
  float*    G2   = W + OF_G2;
  float*    W1e  = W + OF_W1E;
  float*    b1e  = W + OF_B1E;
  float*    b2e  = W + OF_B2E;
  float*    b3e  = W + OF_B3E;
  __bf16*   W2e  = (__bf16*)(W + OF_W2E);
  __bf16*   W3e  = (__bf16*)(W + OF_W3E);
  unsigned* flat = (unsigned*)(W + OF_FLAT);
  __bf16*   h2   = (__bf16*)(W + OF_H2);
  float*    out  = (float*)d_out;

  k_zero<<<(ACC_END + 255) / 256, 256, 0, stream>>>((unsigned*)W, ACC_END);
  k_zero<<<(TOTAL + 255) / 256, 256, 0, stream>>>(flat, TOTAL);
  k_statsx<<<64, 256, 0, stream>>>(x, sx, sxx);
  k_fin1<<<1, 64, 0, stream>>>(w1, b1, g1, be1, sx, sxx, W1e, b1e);
  k_gram1<<<64, 256, 0, stream>>>(x, W1e, b1e, s1, G1);
  k_fin2<<<1, 128, 0, stream>>>(w2, b2, g2, be2, s1, G1, W2e, b2e);
  k_h2<<<256, 256, 0, stream>>>(x, W1e, b1e, W2e, b2e, h2);
  k_gram2<<<64, 256, 0, stream>>>(h2, s2, G2);
  k_fin3<<<4, 256, 0, stream>>>(w3, b3, g3, be3, s2, G2, W3e, b3e);
  k_hist<<<1024, 256, 0, stream>>>(h2, W3e, b3e, flat);
  k_argmax<<<(NGRP + 255) / 256, 256, 0, stream>>>(flat, out);
}
